// TopKRouter_82231443849387
// MI455X (gfx1250) — matrix operand order repaired and checked by reading, not compiled, not run on hardware
//
#include <hip/hip_runtime.h>
#include <hip/hip_bf16.h>
#include <math.h>

// ---------------- problem constants ----------------
#define B_ROWS 32768
#define D_DIM  4096
#define E_DIM  64
#define MT     128   // rows per workgroup
#define KC     64    // K chunk staged in LDS per iteration
#define KP     72    // padded K stride in LDS (elements), 144B rows (16B multiple)

typedef __attribute__((ext_vector_type(16))) __bf16 v16bf;
typedef __attribute__((ext_vector_type(8)))  __bf16 v8bf;
typedef __attribute__((ext_vector_type(2)))  __bf16 v2bf;
typedef __attribute__((ext_vector_type(8)))  float  v8f;

// fp32 -> bf16 via native conversion (RNE); compiler lowers to v_cvt hardware op
__device__ __forceinline__ void st2bf(unsigned short* dst, float a, float b) {
  v2bf p;
  p[0] = (__bf16)a;
  p[1] = (__bf16)b;
  *(v2bf*)dst = p;
}

__device__ __forceinline__ v16bf ldfrag(const unsigned short* p0,
                                        const unsigned short* p1) {
  v8bf lo = *(const v8bf*)p0;               // 16B LDS load
  v8bf hi = *(const v8bf*)p1;               // 16B LDS load
  return __builtin_shufflevector(lo, hi, 0,1,2,3,4,5,6,7,8,9,10,11,12,13,14,15);
}

struct __align__(16) SMem {
  union U {
    struct T {
      unsigned short xT[MT][KP];      // x tile, bf16, [row][k]
      unsigned short wT[E_DIM][KP];   // gate_w tile, bf16, TRANSPOSED [n][k]
      unsigned short nT[E_DIM][KP];   // w_noise tile, bf16, TRANSPOSED [n][k]
    } t;                               // 36864 B
    float logits[MT][E_DIM];           // 32768 B (reused after K loop)
  } u;
  float imp[E_DIM];
  float loadc[E_DIM];
};

__global__ void __launch_bounds__(256)
router_kernel(const float* __restrict__ x, const float* __restrict__ gw,
              const float* __restrict__ gb, const float* __restrict__ wn,
              const float* __restrict__ noise, float* __restrict__ out,
              float* __restrict__ ws)
{
  __shared__ SMem sm;
  const int tid    = threadIdx.x;
  const int wave   = tid >> 5;
  const int lane   = tid & 31;
  const int wgRow0 = blockIdx.x * MT;

  if (tid < E_DIM) { sm.imp[tid] = 0.0f; sm.loadc[tid] = 0.0f; }

  v8f vzero = {};
  v8f accG[4], accN[4];
#pragma unroll
  for (int t = 0; t < 4; ++t) { accG[t] = vzero; accN[t] = vzero; }

  // WMMA fragment addressing (wave32; A and B both use contiguous k = 16h + s)
  const int am    = wave * 16 + (lane & 15); // A: lane -> row
  const int akoff = (lane >> 4) * 16;        // A: lanes 16-31 hold K 16..31
  const int bn    = lane & 15;               // B: lane -> col within 16-tile
  const int bkoff = (lane >> 4) * 16;        // B: lanes 16-31 hold K 16..31

  // weight-transpose staging assignment: thread -> (n, k-quad)
  const int wN  = tid & 63;                  // expert column
  const int wKQ = tid >> 6;                  // base k-quad (0..3)

  for (int kc = 0; kc < D_DIM / KC; ++kc) {
    const int kBase = kc * KC;

    // ---- stage x tile: 128x64 fp32 -> bf16 LDS (8 float4 per thread) ----
#pragma unroll
    for (int j = 0; j < 8; ++j) {
      int idx4 = tid + j * 256;
      int flat = idx4 * 4;
      int row  = flat >> 6, col = flat & 63;
      float4 v = *(const float4*)(x + (size_t)(wgRow0 + row) * D_DIM + kBase + col);
      st2bf(&sm.u.t.xT[row][col],     v.x, v.y);
      st2bf(&sm.u.t.xT[row][col + 2], v.z, v.w);
    }
    // ---- stage weight tiles (64x64 each), transposed to [n][k] ----
    // each thread: column wN, k-quad (wKQ + 4j)*4 -> one contiguous 8B LDS store
#pragma unroll
    for (int j = 0; j < 4; ++j) {
      int kl = (wKQ + j * 4) * 4;            // 0..60, step 4
      const float* gp = gw + (size_t)(kBase + kl) * E_DIM + wN;
      const float* np = wn + (size_t)(kBase + kl) * E_DIM + wN;
      float g0 = gp[0], g1 = gp[E_DIM], g2 = gp[2 * E_DIM], g3 = gp[3 * E_DIM];
      float h0 = np[0], h1 = np[E_DIM], h2 = np[2 * E_DIM], h3 = np[3 * E_DIM];
      st2bf(&sm.u.t.wT[wN][kl],     g0, g1);
      st2bf(&sm.u.t.wT[wN][kl + 2], g2, g3);
      st2bf(&sm.u.t.nT[wN][kl],     h0, h1);
      st2bf(&sm.u.t.nT[wN][kl + 2], h2, h3);
    }
    // ---- prefetch next x chunk (global_prefetch_b8, no counter cost) ----
    if (kc + 1 < D_DIM / KC) {
      const float* nx = x + (size_t)(wgRow0 + (tid >> 1)) * D_DIM
                          + kBase + KC + (tid & 1) * 32;
      __builtin_prefetch(nx, 0, 1);
    }
    __syncthreads();

    // ---- WMMA: 2 k-steps x 4 N-tiles x 2 matrices = 16 wmma / wave ----
#pragma unroll
    for (int kk = 0; kk < KC; kk += 32) {
      v16bf a = ldfrag(&sm.u.t.xT[am][kk + akoff],
                       &sm.u.t.xT[am][kk + akoff + 8]);
#pragma unroll
      for (int t = 0; t < 4; ++t) {
        v16bf bg = ldfrag(&sm.u.t.wT[t * 16 + bn][kk + bkoff],
                          &sm.u.t.wT[t * 16 + bn][kk + bkoff + 8]);
        v16bf bb = ldfrag(&sm.u.t.nT[t * 16 + bn][kk + bkoff],
                          &sm.u.t.nT[t * 16 + bn][kk + bkoff + 8]);
        accG[t] = __builtin_amdgcn_wmma_f32_16x16x32_bf16(
            false, a, false, bg, (short)0, accG[t], false, false);
        accN[t] = __builtin_amdgcn_wmma_f32_16x16x32_bf16(
            false, a, false, bb, (short)0, accN[t], false, false);
      }
    }
    __syncthreads();
  }

  // ---- epilogue: logits = clean + noise * softplus(noise_logits) ----
  // C/D layout: VGPR r, lanes 0-15 -> M=r, N=lane; lanes 16-31 -> M=8+r
  {
    const int laneHi  = lane >> 4;
    const int colBase = lane & 15;
#pragma unroll
    for (int t = 0; t < 4; ++t) {
      int col = t * 16 + colBase;
      float bias = gb[col];
#pragma unroll
      for (int r = 0; r < 8; ++r) {
        int lrow = wave * 16 + laneHi * 8 + r;
        int grow = wgRow0 + lrow;
        float clean = accG[t][r] + bias;
        float nl    = accN[t][r];
        float sp    = fmaxf(nl, 0.0f) + log1pf(expf(-fabsf(nl)));  // softplus
        float lg    = clean + noise[(size_t)grow * E_DIM + col] * sp;
        sm.u.logits[lrow][col] = lg;
      }
    }
  }
  __syncthreads();

  // ---- per-row: top-2, gate softmax, full softmax for importance ----
  if (tid < MT) {
    const float* lr = sm.u.logits[tid];
    float v0 = -3.4e38f, v1 = -3.4e38f;
    int   i0 = 0, i1 = 0;
    for (int e = 0; e < E_DIM; ++e) {
      float l = lr[e];
      if (l > v0)      { v1 = v0; i1 = i0; v0 = l; i0 = e; }
      else if (l > v1) { v1 = l;  i1 = e; }
    }
    float denom = 0.0f;
    for (int e = 0; e < E_DIM; ++e) denom += expf(lr[e] - v0);
    float inv = 1.0f / denom;
    for (int e = 0; e < E_DIM; ++e)
      atomicAdd(&sm.imp[e], expf(lr[e] - v0) * inv);   // ds_add_f32
    atomicAdd(&sm.loadc[i0], 1.0f);
    atomicAdd(&sm.loadc[i1], 1.0f);

    int grow = wgRow0 + tid;
    float g0 = 1.0f / (1.0f + expf(v1 - v0));          // softmax of [v0,v1]
    out[(size_t)grow * 2 + 0] = g0;
    out[(size_t)grow * 2 + 1] = 1.0f - g0;
    out[(size_t)(2 * B_ROWS) + grow * 2 + 0] = (float)i0;
    out[(size_t)(2 * B_ROWS) + grow * 2 + 1] = (float)i1;
  }
  __syncthreads();

  if (tid < E_DIM) {
    atomicAdd(&ws[tid],          sm.imp[tid]);
    atomicAdd(&ws[E_DIM + tid],  sm.loadc[tid]);
  }
}

__global__ void zero_ws_kernel(float* __restrict__ ws) {
  ws[threadIdx.x] = 0.0f;     // 128 threads: imp[64] + load[64]
}

__global__ void finalize_kernel(const float* __restrict__ ws,
                                float* __restrict__ out) {
  __shared__ float buf[E_DIM];
  int t = threadIdx.x;
  buf[t] = ws[t] * ws[E_DIM + t];
  __syncthreads();
  if (t == 0) {
    float s = 0.0f;
    for (int e = 0; e < E_DIM; ++e) s += buf[e];
    // mean over E then * E^2  ==  sum * E
    out[(size_t)4 * B_ROWS] = s * (float)E_DIM;
  }
}

extern "C" void kernel_launch(void* const* d_in, const int* in_sizes, int n_in,
                              void* d_out, int out_size, void* d_ws, size_t ws_size,
                              hipStream_t stream) {
  const float* x     = (const float*)d_in[0];
  const float* gw    = (const float*)d_in[1];
  const float* gb    = (const float*)d_in[2];
  const float* wn    = (const float*)d_in[3];
  const float* noise = (const float*)d_in[4];
  float* out = (float*)d_out;
  float* ws  = (float*)d_ws;

  zero_ws_kernel<<<1, 128, 0, stream>>>(ws);
  router_kernel<<<B_ROWS / MT, 256, 0, stream>>>(x, gw, gb, wn, noise, out, ws);
  finalize_kernel<<<1, E_DIM, 0, stream>>>(ws, out);
}
